// LSHAttention_48765058679201
// MI455X (gfx1250) — matrix operand order repaired and checked by reading, not compiled, not run on hardware
//
#include <hip/hip_runtime.h>
#include <hip/hip_bf16.h>

typedef __attribute__((ext_vector_type(16))) _Float16 v16h;
typedef __attribute__((ext_vector_type(8)))  float    v8f;

#define S_LEN   4096
#define D_DIM   64
#define NH      8
#define NB      64          /* buckets per hash                 */
#define NBK     512         /* total buckets == n_chunks        */
#define NCH     512         /* chunks per batch                 */
#define CS      64          /* chunk (bucket) size              */
#define SORTN   32768       /* NH * S_LEN                       */
#define SK_STR  66          /* f16 LDS row stride (33 banks)    */
#define SKT_STR 130         /* transposed-K row stride (65 bk)  */
#define SP_STR  130         /* probs f16 row stride             */
#define NEG_MAX 3.402823466e+38f

union V16 { v16h v; uint32_t u[8]; };

// ---------------------------------------------------------------------------
// Kernel 1: LSH hashing.  rot = q . rotations, bucket = argmax([rot,-rot]).
// grid (s/256, NH, B), block 256
// ---------------------------------------------------------------------------
__global__ void lsh_hash_kernel(const float* __restrict__ q,
                                const float* __restrict__ rot,
                                int* __restrict__ buckets) {
  __shared__ float rl[64][32];
  const int h = blockIdx.y, b = blockIdx.z, tid = threadIdx.x;
  for (int idx = tid; idx < 64 * 32; idx += 256) {
    int dd = idx >> 5, nn = idx & 31;
    rl[dd][nn] = rot[((size_t)dd * NH + h) * 32 + nn];   // rotations[d][h][n]
  }
  __syncthreads();
  const int t = blockIdx.x * 256 + tid;
  const float* qr = q + ((size_t)b * S_LEN + t) * D_DIM;
  float acc[32];
#pragma unroll
  for (int nn = 0; nn < 32; ++nn) acc[nn] = 0.f;
  for (int dd = 0; dd < 64; ++dd) {
    float qv = qr[dd];
#pragma unroll
    for (int nn = 0; nn < 32; ++nn) acc[nn] += qv * rl[dd][nn];
  }
  float best = acc[0]; int arg = 0;
#pragma unroll
  for (int nn = 1; nn < 32; ++nn) if (acc[nn] > best) { best = acc[nn]; arg = nn; }
#pragma unroll
  for (int nn = 0; nn < 32; ++nn) { float v = -acc[nn]; if (v > best) { best = v; arg = 32 + nn; } }
  buckets[(size_t)b * SORTN + h * S_LEN + t] = arg + h * NB;
}

// ---------------------------------------------------------------------------
// Kernel 2: per-batch bucket histogram + exclusive scan.  grid B, block 256
// ---------------------------------------------------------------------------
__global__ void lsh_hist_kernel(const int* __restrict__ buckets,
                                int* __restrict__ bucket_start) {
  __shared__ int hist[NBK];
  const int b = blockIdx.x, tid = threadIdx.x;
  for (int kk = tid; kk < NBK; kk += 256) hist[kk] = 0;
  __syncthreads();
  for (int i = tid; i < SORTN; i += 256)
    atomicAdd(&hist[buckets[(size_t)b * SORTN + i]], 1);
  __syncthreads();
  if (tid == 0) {
    int run = 0;
    for (int kk = 0; kk < NBK; ++kk) { bucket_start[b * NBK + kk] = run; run += hist[kk]; }
  }
}

// ---------------------------------------------------------------------------
// Kernel 3: deterministic stable counting-sort emission.
// One thread per (batch, bucket); emits in ascending t -> stable permutation.
// grid (B*NBK)/256, block 256
// ---------------------------------------------------------------------------
__global__ void lsh_rank_kernel(const int* __restrict__ buckets,
                                const int* __restrict__ bucket_start,
                                int* __restrict__ sticker,
                                int* __restrict__ undo) {
  const int gid = blockIdx.x * blockDim.x + threadIdx.x;
  const int b = gid >> 9, bkt = gid & (NBK - 1);
  const int h = bkt >> 6;                       // bucket id encodes its hash round
  const int* row = buckets + (size_t)b * SORTN + h * S_LEN;
  int pos = bucket_start[b * NBK + bkt];
  for (int t = 0; t < S_LEN; ++t) {
    if (row[t] == bkt) {
      sticker[(size_t)b * SORTN + pos] = h * S_LEN + t;
      undo[(size_t)b * SORTN + h * S_LEN + t] = pos;
      ++pos;
    }
  }
}

// ---------------------------------------------------------------------------
// Kernel 4: chunked attention. One workgroup per (batch, chunk); 4 waves.
// dots = bq*bk^T via v_wmma_f32_16x16x32_f16, masked softmax in registers,
// bo = probs*bv via WMMA, scattered straight to the unsorted per-hash buffer.
// Keys are stored TRANSPOSED in LDS (skT[d][z]); A fragments are contiguous,
// 4B-aligned f16 pairs (one b32 load each); B pairs need two f16 loads.
// ---------------------------------------------------------------------------
__global__ __launch_bounds__(128)
void lsh_attn_kernel(const float* __restrict__ qp,
                     const float* __restrict__ kp,
                     const float* __restrict__ vp,
                     const int*   __restrict__ buckets,
                     const int*   __restrict__ sticker,
                     const int*   __restrict__ undo,
                     float* __restrict__ logits,
                     float* __restrict__ o_hash) {
  // buf0: sq[64][66] | skT[64][130] during matmul-1; aliased as probs[64][130].
  __shared__ _Float16 buf0[CS * SK_STR + D_DIM * SKT_STR];
  __shared__ _Float16 svl[2 * CS * SK_STR];
  __shared__ int bt_s[2 * CS], bb_s[2 * CS], ii_s[CS];
  __shared__ int L1s[2 * CS][NH], L2s[2 * CS][NH];

  _Float16* sq  = buf0;
  _Float16* skT = buf0 + CS * SK_STR;          // [d][z]: keys transposed

  const int b    = blockIdx.x >> 9;
  const int n    = blockIdx.x & (NCH - 1);
  const int prev = (n + NCH - 1) & (NCH - 1);
  const int tid  = threadIdx.x;

  // ---- Phase 0: gather chunk tiles + metadata into LDS -------------------
  {
    const int z = tid;                                  // 0..127 kv slot
    const int p = (z < CS ? n : prev) * CS + (z & (CS - 1));
    const int i = sticker[(size_t)b * SORTN + p];       // i = h*s + t
    const int t = i & (S_LEN - 1);
    bt_s[z] = t;
    bb_s[z] = buckets[(size_t)b * SORTN + i];
    if (z < CS) ii_s[z] = i;
#pragma unroll
    for (int h = 0; h < NH; ++h) {
      const int bv = buckets[(size_t)b * SORTN + h * S_LEN + t];
      const int ch = undo[(size_t)b * SORTN + h * S_LEN + t] >> 6;
      L1s[z][h] = bv * NCH + ch;
      L2s[z][h] = bv * NCH + ((ch + 1) & (NCH - 1));
    }
    // keys: load, L2-normalize, store TRANSPOSED (skT[d][z])
    {
      const float4* kr = (const float4*)(kp + ((size_t)b * S_LEN + t) * D_DIM);
      float4 reg[16]; float ss = 0.f;
#pragma unroll
      for (int j = 0; j < 16; ++j) {
        reg[j] = kr[j];
        ss += reg[j].x * reg[j].x + reg[j].y * reg[j].y +
              reg[j].z * reg[j].z + reg[j].w * reg[j].w;
      }
      const float rn = rsqrtf(ss);
#pragma unroll
      for (int j = 0; j < 16; ++j) {
        skT[(4 * j + 0) * SKT_STR + z] = (_Float16)(reg[j].x * rn);
        skT[(4 * j + 1) * SKT_STR + z] = (_Float16)(reg[j].y * rn);
        skT[(4 * j + 2) * SKT_STR + z] = (_Float16)(reg[j].z * rn);
        skT[(4 * j + 3) * SKT_STR + z] = (_Float16)(reg[j].w * rn);
      }
    }
    // values (row-major; B fragment pairs for matmul-2 span two rows)
    {
      const float4* vr = (const float4*)(vp + ((size_t)b * S_LEN + t) * D_DIM);
      _Float16* dst = svl + z * SK_STR;
#pragma unroll
      for (int j = 0; j < 16; ++j) {
        float4 x = vr[j];
        dst[4 * j + 0] = (_Float16)x.x; dst[4 * j + 1] = (_Float16)x.y;
        dst[4 * j + 2] = (_Float16)x.z; dst[4 * j + 3] = (_Float16)x.w;
      }
    }
    // queries (current chunk only; row-major for A fragments)
    if (z < CS) {
      const float4* qr = (const float4*)(qp + ((size_t)b * S_LEN + t) * D_DIM);
      _Float16* dst = sq + z * SK_STR;
#pragma unroll
      for (int j = 0; j < 16; ++j) {
        float4 x = qr[j];
        dst[4 * j + 0] = (_Float16)x.x; dst[4 * j + 1] = (_Float16)x.y;
        dst[4 * j + 2] = (_Float16)x.z; dst[4 * j + 3] = (_Float16)x.w;
      }
    }
  }
  __syncthreads();

  const int lane = tid & 31;
  const int wave = tid >> 5;
  const int m0   = wave * 16;            // this wave owns query rows m0..m0+15
  const int mrow = m0 + (lane & 15);
  const int hlf  = lane >> 4;

  // ---- Phase 1: dots[64][128] = bq * bk^T  (K=64, two chained WMMAs) -----
  V16 a0, a1;
#pragma unroll
  for (int v = 0; v < 8; ++v) {
    const int kb = (v < 4 ? 2 * v : 16 + 2 * (v - 4)) + 8 * hlf;  // A K-pair base
    a0.u[v] = *(const uint32_t*)(sq + mrow * SK_STR + kb);
    a1.u[v] = *(const uint32_t*)(sq + mrow * SK_STR + kb + 32);
  }
  const _Float16* bk0 = skT;                            // K rows 0..31
  const _Float16* bk1 = skT + 32 * SKT_STR;             // K rows 32..63
  v8f acc[8];
#pragma unroll
  for (int nt = 0; nt < 8; ++nt) {
    V16 b0, b1;
#pragma unroll
    for (int v = 0; v < 8; ++v) {
      const int nn = nt * 16 + (lane & 15);             // key index (B column)
      const int kb = (v < 4 ? 2 * v : 16 + 2 * (v - 4)) + 8 * hlf;
      b0.v[2 * v]     = bk0[kb * SKT_STR + nn];
      b0.v[2 * v + 1] = bk0[(kb + 1) * SKT_STR + nn];
      b1.v[2 * v]     = bk1[kb * SKT_STR + nn];
      b1.v[2 * v + 1] = bk1[(kb + 1) * SKT_STR + nn];
    }
    v8f c = {};
    c = __builtin_amdgcn_wmma_f32_16x16x32_f16(false, a0.v, false, b0.v, (short)0, c, false, false);
    c = __builtin_amdgcn_wmma_f32_16x16x32_f16(false, a1.v, false, b1.v, (short)0, c, false, false);
    acc[nt] = c;
  }
  __syncthreads();                       // sq/skT now dead -> buf0 becomes probs

  // ---- Phase 2: scale, masks, dup correction, row softmax (registers) ----
  int qt[8], qb[8], ql[8][NH];
#pragma unroll
  for (int r = 0; r < 8; ++r) {
    const int qq = m0 + r + 8 * hlf;     // C-layout row for VGPR r
    qt[r] = bt_s[qq]; qb[r] = bb_s[qq];
#pragma unroll
    for (int h = 0; h < NH; ++h) ql[r][h] = L1s[qq][h];
  }
#pragma unroll
  for (int nt = 0; nt < 8; ++nt) {
    const int z = nt * 16 + (lane & 15);
    const int zt = bt_s[z], zb = bb_s[z];
    int zl1[NH], zl2[NH];
#pragma unroll
    for (int h = 0; h < NH; ++h) { zl1[h] = L1s[z][h]; zl2[h] = L2s[z][h]; }
#pragma unroll
    for (int r = 0; r < 8; ++r) {
      float val = acc[nt][r] * 0.125f;             // d^-0.5
      if (zt == qt[r]) val = -10000.0f;            // self-attention mask
      if (zb != qb[r]) val = -NEG_MAX;             // cross-bucket mask wins
      int cnt = 0;
#pragma unroll
      for (int h = 0; h < NH; ++h)
        cnt += (ql[r][h] == zl1[h]) + (ql[r][h] == zl2[h]);
      val -= logf((float)cnt + 1e-9f);             // duplicate-attn correction
      acc[nt][r] = val;
    }
  }
  float inv_s[8];
  _Float16* probs = buf0;                           // aliased region
#pragma unroll
  for (int r = 0; r < 8; ++r) {
    float rm = -NEG_MAX;
#pragma unroll
    for (int nt = 0; nt < 8; ++nt) rm = fmaxf(rm, acc[nt][r]);
    for (int m = 8; m >= 1; m >>= 1) rm = fmaxf(rm, __shfl_xor(rm, m, 16));
    float rs = 0.f;
#pragma unroll
    for (int nt = 0; nt < 8; ++nt) {
      float e = expf(acc[nt][r] - rm);
      acc[nt][r] = e;
      rs += e;
    }
    for (int m = 8; m >= 1; m >>= 1) rs += __shfl_xor(rs, m, 16);
    inv_s[r] = 1.0f / rs;
    const int qq = m0 + r + 8 * hlf;
    if ((lane & 15) == 0)
      logits[(size_t)b * SORTN + ii_s[qq]] = logf(rs) + rm;   // lse, unsorted
#pragma unroll
    for (int nt = 0; nt < 8; ++nt) {
      const int z = nt * 16 + (lane & 15);
      probs[qq * SP_STR + z] = (_Float16)acc[nt][r];          // unnormalized
    }
  }
  __syncthreads();

  // ---- Phase 3: bo[64][64] = probs[64][128] * bv[128][64] via WMMA -------
#pragma unroll
  for (int nt2 = 0; nt2 < 4; ++nt2) {
    v8f c = {};
#pragma unroll
    for (int kb = 0; kb < 128; kb += 32) {
      V16 a, bm;
#pragma unroll
      for (int v = 0; v < 8; ++v) {
        const int kk = kb + (v < 4 ? 2 * v : 16 + 2 * (v - 4)) + 8 * hlf;
        a.u[v] = *(const uint32_t*)(probs + mrow * SP_STR + kk);
        const int nn = nt2 * 16 + (lane & 15);
        bm.v[2 * v]     = svl[kk * SK_STR + nn];
        bm.v[2 * v + 1] = svl[(kk + 1) * SK_STR + nn];
      }
      c = __builtin_amdgcn_wmma_f32_16x16x32_f16(false, a.v, false, bm.v, (short)0, c, false, false);
    }
#pragma unroll
    for (int r = 0; r < 8; ++r) {
      const int row = m0 + r + 8 * hlf;
      const int col = nt2 * 16 + (lane & 15);
      o_hash[((size_t)b * SORTN + ii_s[row]) * D_DIM + col] = c[r] * inv_s[r];
    }
  }
}

// ---------------------------------------------------------------------------
// Kernel 5: combine hash rounds: out = sum_h o_h * softmax_h(lse_h).
// grid B*S, block 64 (one thread per d)
// ---------------------------------------------------------------------------
__global__ void lsh_combine_kernel(const float* __restrict__ o_hash,
                                   const float* __restrict__ logits,
                                   float* __restrict__ out) {
  const int b = blockIdx.x >> 12, t = blockIdx.x & (S_LEN - 1), d = threadIdx.x;
  float l[NH], m = -NEG_MAX;
#pragma unroll
  for (int h = 0; h < NH; ++h) {
    l[h] = logits[(size_t)b * SORTN + h * S_LEN + t];
    m = fmaxf(m, l[h]);
  }
  float W = 0.f, accv = 0.f;
#pragma unroll
  for (int h = 0; h < NH; ++h) {
    float w = expf(l[h] - m);
    W += w;
    accv += w * o_hash[((size_t)b * SORTN + h * S_LEN + t) * D_DIM + d];
  }
  out[((size_t)b * S_LEN + t) * D_DIM + d] = accv / W;
}

// ---------------------------------------------------------------------------
extern "C" void kernel_launch(void* const* d_in, const int* in_sizes, int n_in,
                              void* d_out, int out_size, void* d_ws, size_t ws_size,
                              hipStream_t stream) {
  const float* q   = (const float*)d_in[0];
  const float* k   = (const float*)d_in[1];
  const float* v   = (const float*)d_in[2];
  const float* rot = (const float*)d_in[3];
  float* out = (float*)d_out;

  const int B = in_sizes[0] / (S_LEN * D_DIM);   // 4

  char* ws = (char*)d_ws;
  size_t off = 0;
  int* buckets = (int*)(ws + off); off += (size_t)B * SORTN * sizeof(int);
  int* sticker = (int*)(ws + off); off += (size_t)B * SORTN * sizeof(int);
  int* undo    = (int*)(ws + off); off += (size_t)B * SORTN * sizeof(int);
  int* bstart  = (int*)(ws + off); off += (size_t)B * NBK   * sizeof(int);
  float* logits = (float*)(ws + off); off += (size_t)B * SORTN * sizeof(float);
  float* o_hash = (float*)(ws + off); off += (size_t)B * SORTN * D_DIM * sizeof(float);
  (void)ws_size; (void)n_in; (void)out_size;

  lsh_hash_kernel<<<dim3(S_LEN / 256, NH, B), 256, 0, stream>>>(q, rot, buckets);
  lsh_hist_kernel<<<B, 256, 0, stream>>>(buckets, bstart);
  lsh_rank_kernel<<<(B * NBK) / 256, 256, 0, stream>>>(buckets, bstart, sticker, undo);
  lsh_attn_kernel<<<B * NCH, 128, 0, stream>>>(q, k, v, buckets, sticker, undo,
                                               logits, o_hash);
  lsh_combine_kernel<<<B * S_LEN, D_DIM, 0, stream>>>(o_hash, logits, out);
}
